// GraphConvNet_89275190215567
// MI455X (gfx1250) — compile-verified, matrix operand order checked by reading, not run on hardware
//
#include <hip/hip_runtime.h>

typedef __attribute__((ext_vector_type(16))) _Float16 v16h;
typedef __attribute__((ext_vector_type(8)))  float    v8f;

#define NREG   90
#define NNODE  91
#define NPAIR  4005
#define H1     256
#define ASTRIDE 96          // padded row stride for Afull16 (96x96, zero-padded)
#define MT     6            // M tiles (96 rows)
#define NT     16           // N tiles (256 cols)
#define KT     3            // K steps (96 = 3*32)
#define BP_ELEMS (NT*KT*32*16)   // 24576 halves, 48 KB packed W1 fragments
#define AP_ELEMS (MT*KT*32*16)   // 9216 halves,  18 KB packed A fragments

// K index within a 16x32 f16 WMMA operand fragment, per CDNA5 ISA layout.
__device__ __forceinline__ int frag_k(int lane, int e) {
    int v = e >> 1, p = e & 1;
    int k = (v < 4) ? (2 * v + p) : (16 + 2 * (v - 4) + p);
    return k + ((lane >= 16) ? 8 : 0);
}

// ---------------------------------------------------------------------------
// Pre-pass: pack W1 [256 x 91] into B-fragment layout (32x16 f16 per frag),
// fragment f = (ni*3 + ki)*32 + lane, 16 halves per lane. Runs once; result
// (48 KB) is shared by all 4096 graph blocks and stays L2-resident.
// ---------------------------------------------------------------------------
__global__ __launch_bounds__(256) void pack_w1_kernel(const float* __restrict__ W1,
                                                      _Float16* __restrict__ Bp) {
    int idx = blockIdx.x * 256 + threadIdx.x;
    if (idx >= BP_ELEMS) return;
    int e    = idx & 15;
    int lane = (idx >> 4) & 31;
    int fi   = idx >> 9;          // ni*3 + ki
    int ki   = fi % 3;
    int ni   = fi / 3;
    int j = ki * 32 + frag_k(lane, e);   // K dim  (region index)
    int h = ni * 16 + (lane & 15);       // N dim  (feature index)
    float val = (j < NNODE) ? W1[h * NNODE + j] : 0.0f;
    Bp[idx] = (_Float16)val;
}

// ---------------------------------------------------------------------------
// Main kernel: one 256-thread block per graph.
// ---------------------------------------------------------------------------
__global__ __launch_bounds__(256) void graphconv_kernel(
        const float* __restrict__ coh,
        const float* __restrict__ W2,
        const float* __restrict__ Wo1, const float* __restrict__ bo1,
        const float* __restrict__ Wo2, const float* __restrict__ bo2,
        const _Float16* __restrict__ Bp,   // packed W1 fragments (global, L2-hot)
        float* __restrict__ out) {

    __shared__ __align__(16) _Float16 Afull16[ASTRIDE * ASTRIDE]; // A + I (unnorm, zero-padded)
    __shared__ __align__(32) _Float16 Ap[AP_ELEMS];               // packed A fragments
    __shared__ float rowsum[96];
    __shared__ float dinvs[96];                   // zero-padded past 90
    __shared__ __align__(16) float a90s[96];      // normalized supernode row, zero-padded
    __shared__ float ys[H1];         // y_h = sum_j a90[j] * relu(X1[j,h])
    __shared__ float zs[H1];         // pooled @ W2^T
    __shared__ float fc1s[128];

    const int tid  = threadIdx.x;
    const int lane = tid & 31;
    const int wave = tid >> 5;
    const int b    = blockIdx.x;
    const float* cohb = coh + (size_t)b * NPAIR;

    // ---- phase 0: zero A (16B stores), init rowsums, zero ys ------------
    {
        uint4 zz = {0u, 0u, 0u, 0u};
        uint4* az = (uint4*)Afull16;
        for (int q = tid; q < (ASTRIDE * ASTRIDE * 2) / 16; q += 256) az[q] = zz;
    }
    if (tid < 96)  rowsum[tid] = (tid < NREG) ? 2.0f : 91.0f; // diag + supernode link
    if (tid < H1)  ys[tid] = 0.0f;
    __syncthreads();

    // ---- phase 1: scatter coherence into A, accumulate row sums ---------
    for (int q = tid; q < NREG * NREG; q += 256) {
        int i = q / NREG, j = q % NREG;
        if (i < j) {
            int p = i * 89 - (i * (i - 1)) / 2 + (j - i - 1);
            float v = cohb[p];
            _Float16 hv = (_Float16)v;
            Afull16[i * ASTRIDE + j] = hv;
            Afull16[j * ASTRIDE + i] = hv;
            atomicAdd(&rowsum[i], v);
            atomicAdd(&rowsum[j], v);
        }
    }
    if (tid < NNODE) {               // supernode row/col + identity diagonal
        Afull16[NREG * ASTRIDE + tid] = (_Float16)1.0f;
        if (tid < NREG) {
            Afull16[tid * ASTRIDE + NREG] = (_Float16)1.0f;
            Afull16[tid * ASTRIDE + tid]  = (_Float16)1.0f;
        }
    }
    __syncthreads();

    // ---- phase 2: dinv = rsqrt(rowsum), zero-padded ---------------------
    if (tid < 96) dinvs[tid] = (tid < NNODE) ? __frsqrt_rn(rowsum[tid]) : 0.0f;
    __syncthreads();

    // ---- phase 3: pack normalized A into A-fragment layout (branch-free)
    for (int idx = tid; idx < AP_ELEMS; idx += 256) {
        int e  = idx & 15;
        int ln = (idx >> 4) & 31;
        int fi = idx >> 9;           // mi*3 + ki
        int ki = fi % 3;
        int mi = fi / 3;
        int m = mi * 16 + (ln & 15);
        int j = ki * 32 + frag_k(ln, e);
        // padding rows/cols have dinv == 0 -> value 0, no bounds check
        float val = (float)Afull16[m * ASTRIDE + j] * dinvs[m] * dinvs[j];
        Ap[idx] = (_Float16)val;
    }
    if (tid < 96)                    // A[90,j]==1 for all j<=90; zero-pad rest
        a90s[tid] = (tid < NNODE) ? dinvs[NREG] * dinvs[tid] : 0.0f;
    __syncthreads();

    // ---- phase 4: WMMA GEMM X1 = relu(A_hat @ W1^T), fused supernode
    //      contraction y_h += a90[m] * relu(X1[m,h]) via LDS atomics -------
    const v16h* ApV = (const v16h*)Ap;
    const v16h* BpV = (const v16h*)Bp;
    const int n      = lane & 15;
    const int mbase0 = (lane >> 4) << 3;      // 0 or 8

    for (int t = wave; t < MT * NT; t += 8) { // 12 tiles per wave
        int mi = t >> 4;                      // t / 16
        int ni = t & 15;                      // t % 16
        v8f acc = {};
        #pragma unroll
        for (int ki = 0; ki < KT; ++ki) {
            v16h a  = ApV[(mi * 3 + ki) * 32 + lane];   // 2x ds_load_b128
            v16h bf = BpV[(ni * 3 + ki) * 32 + lane];   // 2x global_load_b128 (L2)
            acc = __builtin_amdgcn_wmma_f32_16x16x32_f16(
                    false, a, false, bf, (short)0, acc, false, false);
        }
        // branch-free weighted relu-reduction: 2x ds_load_b128 of a90 weights
        float4 aw0 = *(const float4*)&a90s[mi * 16 + mbase0];
        float4 aw1 = *(const float4*)&a90s[mi * 16 + mbase0 + 4];
        float s;
        s  = aw0.x * fmaxf(acc[0], 0.0f);
        s += aw0.y * fmaxf(acc[1], 0.0f);
        s += aw0.z * fmaxf(acc[2], 0.0f);
        s += aw0.w * fmaxf(acc[3], 0.0f);
        s += aw1.x * fmaxf(acc[4], 0.0f);
        s += aw1.y * fmaxf(acc[5], 0.0f);
        s += aw1.z * fmaxf(acc[6], 0.0f);
        s += aw1.w * fmaxf(acc[7], 0.0f);
        atomicAdd(&ys[ni * 16 + n], s);                 // ds_add_f32
    }
    __syncthreads();

    // ---- phase 5: pooled = y @ W2^T  (256 dots of length 256) -----------
    {
        const float4* w2r = (const float4*)(W2 + tid * H1);
        float acc = 0.0f;
        #pragma unroll 8
        for (int q = 0; q < H1 / 4; ++q) {
            float4 w = w2r[q];
            acc += w.x * ys[4 * q] + w.y * ys[4 * q + 1]
                 + w.z * ys[4 * q + 2] + w.w * ys[4 * q + 3];
        }
        zs[tid] = acc;
    }
    __syncthreads();

    // ---- phase 6: fc1 = relu(z @ Wo1^T + bo1) ---------------------------
    if (tid < 128) {
        const float4* wr = (const float4*)(Wo1 + tid * H1);
        float acc = bo1[tid];
        #pragma unroll 8
        for (int q = 0; q < H1 / 4; ++q) {
            float4 w = wr[q];
            acc += w.x * zs[4 * q] + w.y * zs[4 * q + 1]
                 + w.z * zs[4 * q + 2] + w.w * zs[4 * q + 3];
        }
        fc1s[tid] = fmaxf(acc, 0.0f);
    }
    __syncthreads();

    // ---- phase 7: out = fc1 @ Wo2^T + bo2 -------------------------------
    if (tid < 2) {
        const float4* wr = (const float4*)(Wo2 + tid * 128);
        float acc = bo2[tid];
        #pragma unroll
        for (int q = 0; q < 128 / 4; ++q) {
            float4 w = wr[q];
            acc += w.x * fc1s[4 * q] + w.y * fc1s[4 * q + 1]
                 + w.z * fc1s[4 * q + 2] + w.w * fc1s[4 * q + 3];
        }
        out[(size_t)b * 2 + tid] = acc;
    }
}

extern "C" void kernel_launch(void* const* d_in, const int* in_sizes, int n_in,
                              void* d_out, int out_size, void* d_ws, size_t ws_size,
                              hipStream_t stream) {
    const float* coh = (const float*)d_in[0];
    const float* W1  = (const float*)d_in[1];
    const float* W2  = (const float*)d_in[2];
    const float* Wo1 = (const float*)d_in[3];
    const float* bo1 = (const float*)d_in[4];
    const float* Wo2 = (const float*)d_in[5];
    const float* bo2 = (const float*)d_in[6];
    float* out = (float*)d_out;

    _Float16* Bp = (_Float16*)d_ws;   // 48 KB packed W1 fragments
    int Bn = in_sizes[0] / NPAIR;     // batch size (4096)

    pack_w1_kernel<<<(BP_ELEMS + 255) / 256, 256, 0, stream>>>(W1, Bp);
    graphconv_kernel<<<Bn, 256, 0, stream>>>(coh, W2, Wo1, bo1, Wo2, bo2, Bp, out);
}
